// EFN_15427522527435
// MI455X (gfx1250) — matrix operand order repaired and checked by reading, not compiled, not run on hardware
//
#include <hip/hip_runtime.h>

// ---------------------------------------------------------------------------
// EFN edge-MLP for MI455X (gfx1250), bf16 WMMA path.
//   out[n,64] = segment_sum( relu([h_i, h_j-h_i] @ W1 + b1) @ W2 + b2, dst )
// Computed transposed so gathered edge features are the WMMA B operand:
//   H1^T = W1^T x m_in^T ; D2^T = W2^T x relu(H1^T) ; scatter D2^T columns.
// ---------------------------------------------------------------------------

typedef __bf16 bf16x16 __attribute__((ext_vector_type(16)));
typedef float  f32x8   __attribute__((ext_vector_type(8)));
typedef float  f32x4   __attribute__((ext_vector_type(4)));

static __device__ __forceinline__ unsigned short f2bf(float f) {
  unsigned u = __builtin_bit_cast(unsigned, f);
  return (unsigned short)((u + 0x7FFFu + ((u >> 16) & 1u)) >> 16);   // RNE
}
static __device__ __forceinline__ unsigned pk2(float a, float b) {
  return (unsigned)f2bf(a) | ((unsigned)f2bf(b) << 16);
}

// ---------------------------------------------------------------------------
// Pre-pack W1 (96x96, KxM) and W2 (96x64, KxM) into WMMA A-fragment layout:
//   frag(mb,kb): lane L holds row m = 16*mb + (L&15); halves 0..7  -> K = 32*kb + 16*(L>>4) + j
//                                                     halves 8..15 -> K = 32*kb + 16*(L>>4) + 8 + j
// Stored as [frag][lane][16] contiguous bf16 -> one 32B load per lane.
// ---------------------------------------------------------------------------
__global__ void efn_pack_weights(const float* __restrict__ W1,
                                 const float* __restrict__ W2,
                                 unsigned short* __restrict__ ws) {
  const int t = blockIdx.x * blockDim.x + threadIdx.x;
  if (t < 576) {                                     // W1: 6 mb x 3 kb x 32 lanes
    const int L = t & 31, frag = t >> 5;
    const int mb = frag / 3, kb = frag % 3;
    const int m = mb * 16 + (L & 15);
    const int kh = L >> 4;
    unsigned short* o = ws + t * 16;
#pragma unroll
    for (int j = 0; j < 8; ++j) {
      o[j]     = f2bf(W1[(kb * 32 + kh * 16 + j) * 96 + m]);
      o[8 + j] = f2bf(W1[(kb * 32 + kh * 16 + 8 + j) * 96 + m]);
    }
  } else if (t < 960) {                              // W2: 4 mb x 3 kb x 32 lanes
    const int t2 = t - 576;
    const int L = t2 & 31, frag = t2 >> 5;
    const int mb = frag / 3, kb = frag % 3;
    const int m = mb * 16 + (L & 15);
    const int kh = L >> 4;
    unsigned short* o = ws + 9216 + t2 * 16;
#pragma unroll
    for (int j = 0; j < 8; ++j) {
      o[j]     = f2bf(W2[(kb * 32 + kh * 16 + j) * 64 + m]);
      o[8 + j] = f2bf(W2[(kb * 32 + kh * 16 + 8 + j) * 64 + m]);
    }
  }
}

__global__ void efn_zero(float* __restrict__ p, int n) {
  int i = blockIdx.x * blockDim.x + threadIdx.x;
  if (i < n) p[i] = 0.f;
}

// ---------------------------------------------------------------------------
// Main kernel: 8 waves/block, each wave owns one 16-edge tile start-to-finish.
// ---------------------------------------------------------------------------
__global__ void __launch_bounds__(256) efn_edge_mlp(
    const float* __restrict__ x,
    const float* __restrict__ scalars,
    const float* __restrict__ b1,
    const float* __restrict__ b2,
    const long long* __restrict__ eidx,       // int64 [2][E]
    const unsigned short* __restrict__ w1A,   // packed bf16 A fragments
    const unsigned short* __restrict__ w2A,
    float* __restrict__ out,
    int nEdges) {
  constexpr int STRIDE = 112;                 // halves per LDS row (224B, 32B aligned)
  __shared__ unsigned short ldsIn[8 * 16 * STRIDE];   // m_in^T tiles  (edge x 96 feat)
  __shared__ unsigned short ldsH1[8 * 16 * STRIDE];   // relu(H1)^T tiles (edge x 96 hidden)

  const int lane  = threadIdx.x & 31;
  const int wave  = threadIdx.x >> 5;
  const int row   = lane & 15;                // edge within tile / B column
  const int khalf = lane >> 4;                // lane half-group
  const int tile  = blockIdx.x * 8 + wave;
  const int base  = tile * 16;

  unsigned short* bufIn = &ldsIn[wave * 16 * STRIDE];
  unsigned short* bufH1 = &ldsH1[wave * 16 * STRIDE];

  const int  eRaw  = base + row;
  const bool valid = eRaw < nEdges;
  const int  e     = valid ? eRaw : (nEdges - 1);     // clamp; keeps barriers uniform

  // ---- Gather phase: build 16x96 bf16 m_in tile. Two lanes per edge. ----
  if (khalf == 0) {
    // cols 0..47: h_i = [x[dst], scalars]
    const long long d = eidx[(long long)nEdges + e];
    const float* xr = x + d * 32ll;
    unsigned short* wr = bufIn + row * STRIDE;
#pragma unroll
    for (int q = 0; q < 4; ++q) {
      f32x4 a = *(const f32x4*)(xr + q * 8);
      f32x4 c = *(const f32x4*)(xr + q * 8 + 4);
      uint4 p;
      p.x = pk2(a.x, a.y); p.y = pk2(a.z, a.w);
      p.z = pk2(c.x, c.y); p.w = pk2(c.z, c.w);
      *(uint4*)(wr + q * 8) = p;
    }
#pragma unroll
    for (int q = 0; q < 2; ++q) {
      f32x4 a = *(const f32x4*)(scalars + q * 8);
      f32x4 c = *(const f32x4*)(scalars + q * 8 + 4);
      uint4 p;
      p.x = pk2(a.x, a.y); p.y = pk2(a.z, a.w);
      p.z = pk2(c.x, c.y); p.w = pk2(c.z, c.w);
      *(uint4*)(wr + 32 + q * 8) = p;
    }
  } else {
    // cols 48..95: h_j - h_i  (scalar dims cancel -> zeros in cols 80..95)
    const long long s = eidx[e];
    const long long d = eidx[(long long)nEdges + e];
    const float* xs = x + s * 32ll;
    const float* xd = x + d * 32ll;
    unsigned short* wr = bufIn + row * STRIDE + 48;
#pragma unroll
    for (int q = 0; q < 4; ++q) {
      f32x4 a  = *(const f32x4*)(xs + q * 8);
      f32x4 c  = *(const f32x4*)(xs + q * 8 + 4);
      f32x4 da = *(const f32x4*)(xd + q * 8);
      f32x4 dc = *(const f32x4*)(xd + q * 8 + 4);
      uint4 p;
      p.x = pk2(a.x - da.x, a.y - da.y); p.y = pk2(a.z - da.z, a.w - da.w);
      p.z = pk2(c.x - dc.x, c.y - dc.y); p.w = pk2(c.z - dc.z, c.w - dc.w);
      *(uint4*)(wr + q * 8) = p;
    }
    uint4 z; z.x = z.y = z.z = z.w = 0u;
    *(uint4*)(wr + 32) = z;
    *(uint4*)(wr + 40) = z;
  }
  __syncthreads();

  // ---- GEMM1: H1^T = W1^T x m_in^T ; acc tiles mb=0..5 (hidden x edge), +b1 ----
  f32x8 acc1[6];
#pragma unroll
  for (int mb = 0; mb < 6; ++mb)
    acc1[mb] = *(const f32x8*)(b1 + mb * 16 + khalf * 8);   // C layout: M = 16mb+8kh+r
#pragma unroll
  for (int kb = 0; kb < 3; ++kb) {
    bf16x16 bfrag = *(const bf16x16*)(bufIn + row * STRIDE + kb * 32 + khalf * 16);
#pragma unroll
    for (int mb = 0; mb < 6; ++mb) {
      bf16x16 afrag = *(const bf16x16*)(w1A + ((mb * 3 + kb) * 32 + lane) * 16);
      acc1[mb] = __builtin_amdgcn_wmma_f32_16x16x32_bf16(
          false, afrag, false, bfrag, (short)0, acc1[mb], false, false);
    }
  }

  // ---- ReLU + bf16, store back to LDS (one ds_store_b128 per tile per lane) ----
#pragma unroll
  for (int mb = 0; mb < 6; ++mb) {
    f32x8 a = acc1[mb];
    uint4 p;
    p.x = pk2(__builtin_fmaxf(a[0], 0.f), __builtin_fmaxf(a[1], 0.f));
    p.y = pk2(__builtin_fmaxf(a[2], 0.f), __builtin_fmaxf(a[3], 0.f));
    p.z = pk2(__builtin_fmaxf(a[4], 0.f), __builtin_fmaxf(a[5], 0.f));
    p.w = pk2(__builtin_fmaxf(a[6], 0.f), __builtin_fmaxf(a[7], 0.f));
    *(uint4*)(bufH1 + row * STRIDE + mb * 16 + khalf * 8) = p;   // hidden-contiguous
  }
  __syncthreads();

  // ---- GEMM2: D2^T = W2^T x relu(H1^T) ; acc tiles mb2=0..3 (out x edge), +b2 ----
  f32x8 acc2[4];
#pragma unroll
  for (int mb = 0; mb < 4; ++mb)
    acc2[mb] = *(const f32x8*)(b2 + mb * 16 + khalf * 8);
#pragma unroll
  for (int kb = 0; kb < 3; ++kb) {
    bf16x16 bfrag = *(const bf16x16*)(bufH1 + row * STRIDE + kb * 32 + khalf * 16);
#pragma unroll
    for (int mb = 0; mb < 4; ++mb) {
      bf16x16 afrag = *(const bf16x16*)(w2A + ((mb * 3 + kb) * 32 + lane) * 16);
      acc2[mb] = __builtin_amdgcn_wmma_f32_16x16x32_bf16(
          false, afrag, false, bfrag, (short)0, acc2[mb], false, false);
    }
  }

  // ---- Scatter-add: lane holds out dims 16*mb+8*khalf+r of edge `row` ----
  if (valid) {
    const long long d = eidx[(long long)nEdges + eRaw];
    float* op = out + d * 64ll + khalf * 8;
#pragma unroll
    for (int mb = 0; mb < 4; ++mb)
#pragma unroll
      for (int r = 0; r < 8; ++r)
        unsafeAtomicAdd(op + mb * 16 + r, acc2[mb][r]);   // global_atomic_add_f32
  }
}

// ---------------------------------------------------------------------------
extern "C" void kernel_launch(void* const* d_in, const int* in_sizes, int n_in,
                              void* d_out, int out_size, void* d_ws, size_t ws_size,
                              hipStream_t stream) {
  (void)n_in; (void)ws_size;
  const float*     x    = (const float*)d_in[0];      // [N,32]
  const float*     scal = (const float*)d_in[1];      // [1,16]
  const float*     W1   = (const float*)d_in[2];      // [96,96]
  const float*     b1   = (const float*)d_in[3];      // [96]
  const float*     W2   = (const float*)d_in[4];      // [96,64]
  const float*     b2   = (const float*)d_in[5];      // [64]
  const long long* eidx = (const long long*)d_in[6];  // int64 [2,E]
  float*           out  = (float*)d_out;
  unsigned short*  wsp  = (unsigned short*)d_ws;      // 9216 + 6144 bf16 = 30720 B

  const int nEdges = in_sizes[6] / 2;

  efn_pack_weights<<<4, 256, 0, stream>>>(W1, W2, wsp);
  efn_zero<<<(out_size + 255) / 256, 256, 0, stream>>>(out, out_size);

  const int tiles  = (nEdges + 15) / 16;   // 50000
  const int blocks = (tiles + 7) / 8;      // 6250 blocks x 8 waves = 1 tile/wave
  efn_edge_mlp<<<blocks, 256, 0, stream>>>(x, scal, b1, b2, eidx,
                                           wsp, wsp + 9216, out, nEdges);
}
